// MultiheadAttention_80212809220527
// MI455X (gfx1250) — compile-verified, matrix operand order checked by reading, not run on hardware
//
#include <hip/hip_runtime.h>

// ---------------------------------------------------------------------------
// Problem constants (B=4, T=2048, D=1024, H=16, HD=64)
// ---------------------------------------------------------------------------
#define BB 4
#define TT 2048
#define DD 1024
#define HH 16
#define HD 64
#define MM (BB * TT)       // 8192 rows
#define N_QKV (3 * DD)     // 3072

typedef __attribute__((ext_vector_type(16))) __bf16 v16bf;
typedef __attribute__((ext_vector_type(8)))  __bf16 v8bf;
typedef __attribute__((ext_vector_type(8)))  float  v8f;

// fold softmax scale (1/sqrt(HD)=0.125) and log2(e) into q at rope time,
// so the softmax inner loop is pure exp2 (v_exp_f32).
#define QSCALE (0.125f * 1.44269504088896340736f)

__device__ __forceinline__ __bf16 f2bf(float f) {
  unsigned u = __builtin_bit_cast(unsigned, f);
  unsigned r = u + 0x7FFFu + ((u >> 16) & 1u);   // round-to-nearest-even
  return __builtin_bit_cast(__bf16, (unsigned short)(r >> 16));
}
__device__ __forceinline__ float bf2f(__bf16 b) {
  unsigned u = ((unsigned)__builtin_bit_cast(unsigned short, b)) << 16;
  return __builtin_bit_cast(float, u);
}

// Load a 16x32 bf16 WMMA A-fragment (ISA 16-bit A layout, wave32):
//   lane<16 : row=lane,    K = {koff..koff+7, 16+koff..16+koff+7}, koff=0
//   lane>=16: row=lane-16, same with koff=8
// Works identically for B-fragments of K^T-style operands (row-major [N,K]).
__device__ __forceinline__ v16bf load_frag(const __bf16* __restrict__ base,
                                           int ld, int lane) {
  int m    = lane & 15;
  int koff = (lane >> 4) << 3;               // 0 or 8
  const __bf16* p = base + (size_t)m * ld + koff;
  v8bf lo = *(const v8bf*)(p);
  v8bf hi = *(const v8bf*)(p + 16);
  return __builtin_shufflevector(lo, hi, 0,1,2,3,4,5,6,7,8,9,10,11,12,13,14,15);
}

__device__ __forceinline__ v8f wmma_bf16(v16bf a, v16bf b, v8f c) {
  return __builtin_amdgcn_wmma_f32_16x16x32_bf16(
      /*neg_a=*/false, a, /*neg_b=*/false, b,
      /*c_mod=*/(short)0, c, /*reuse_a=*/false, /*reuse_b=*/false);
}

// gfx1250 async global->LDS copy (ASYNCcnt-tracked), per-lane 16 bytes.
__device__ __forceinline__ void async_ld_b128(unsigned lds_addr,
                                              const void* gaddr) {
  asm volatile("global_load_async_to_lds_b128 %0, %1, off"
               :: "v"(lds_addr), "v"(gaddr)
               : "memory");
}
__device__ __forceinline__ void wait_async0() {
  asm volatile("s_wait_asynccnt 0x0" ::: "memory");
}
__device__ __forceinline__ unsigned lds_off(const void* p) {
  // generic pointer to a __shared__ object: low 32 bits are the LDS offset
  return (unsigned)(unsigned long long)p;
}

// ---------------------------------------------------------------------------
// fp32 -> bf16 elementwise convert
// ---------------------------------------------------------------------------
__global__ void f32_to_bf16_kernel(const float* __restrict__ src,
                                   __bf16* __restrict__ dst, int n) {
  int i = blockIdx.x * blockDim.x + threadIdx.x;
  if (i < n) dst[i] = f2bf(src[i]);
}

// ---------------------------------------------------------------------------
// GEMM: out[M,N] = A[M,K=DD] * W[N,K=DD]^T + bias[N]
// One wave computes a 64x32 tile (4x2 WMMA accumulators). Fragments come
// straight from global (operands are L2-resident on the 192 MB L2); the
// K-loop is register double-buffered so loads overlap WMMA issue.
// ---------------------------------------------------------------------------
template <bool OUT_BF16>
__global__ __launch_bounds__(256) void gemm_bf16_kernel(
    const __bf16* __restrict__ A, const __bf16* __restrict__ W,
    const float* __restrict__ bias, void* __restrict__ out,
    int M, int N) {
  int lane = threadIdx.x & 31;
  int wid  = blockIdx.x * (blockDim.x >> 5) + (threadIdx.x >> 5);
  int tilesM = M >> 6;
  int tm = (wid % tilesM) << 6;
  int tn = (wid / tilesM) << 5;
  if (tn >= N) return;

  v8f acc[4][2];
  const v8f vzero = {0.f, 0.f, 0.f, 0.f, 0.f, 0.f, 0.f, 0.f};
#pragma unroll
  for (int i = 0; i < 4; ++i)
#pragma unroll
    for (int j = 0; j < 2; ++j) acc[i][j] = vzero;

  v16bf af[2][4], bf[2][2];
  auto loadAB = [&](int k0, int s) {
#pragma unroll
    for (int i = 0; i < 4; ++i)
      af[s][i] = load_frag(A + (size_t)(tm + 16 * i) * DD + k0, DD, lane);
#pragma unroll
    for (int j = 0; j < 2; ++j)
      bf[s][j] = load_frag(W + (size_t)(tn + 16 * j) * DD + k0, DD, lane);
  };

  loadAB(0, 0);
  for (int k0 = 0; k0 < DD; k0 += 32) {
    int s = (k0 >> 5) & 1;
    if (k0 + 32 < DD) loadAB(k0 + 32, s ^ 1);   // prefetch next chunk
#pragma unroll
    for (int i = 0; i < 4; ++i)
#pragma unroll
      for (int j = 0; j < 2; ++j)
        acc[i][j] = wmma_bf16(af[s][i], bf[s][j], acc[i][j]);
  }

  int half = lane >> 4, col0 = lane & 15;
#pragma unroll
  for (int i = 0; i < 4; ++i) {
#pragma unroll
    for (int j = 0; j < 2; ++j) {
      int col = tn + 16 * j + col0;
      float bv = bias[col];
#pragma unroll
      for (int v = 0; v < 8; ++v) {
        int row = tm + 16 * i + v + 8 * half;
        float val = acc[i][j][v] + bv;
        if constexpr (OUT_BF16)
          ((__bf16*)out)[(size_t)row * N + col] = f2bf(val);
        else
          ((float*)out)[(size_t)row * N + col] = val;
      }
    }
  }
}

// ---------------------------------------------------------------------------
// RoPE + head split: qkv[M, 3D] bf16 ->
//   qh[B,H,T,HD] (scaled by QSCALE), kh[B,H,T,HD], vT[B,H,HD,T]
// ---------------------------------------------------------------------------
__global__ __launch_bounds__(256) void rope_split_kernel(
    const __bf16* __restrict__ qkv, const float* __restrict__ fcos,
    const float* __restrict__ fsin, __bf16* __restrict__ qh,
    __bf16* __restrict__ kh, __bf16* __restrict__ vT) {
  int idx = blockIdx.x * blockDim.x + threadIdx.x;   // B*T*H*(HD/2) = 2^22
  int i = idx & 31;
  int h = (idx >> 5) & (HH - 1);
  int t = (idx >> 9) & (TT - 1);
  int b = idx >> 20;

  size_t row = (size_t)(b * TT + t) * N_QKV;
  int c0 = h * HD + 2 * i;
  float q0 = bf2f(qkv[row + c0]),          q1 = bf2f(qkv[row + c0 + 1]);
  float k0 = bf2f(qkv[row + DD + c0]),     k1 = bf2f(qkv[row + DD + c0 + 1]);
  float v0 = bf2f(qkv[row + 2 * DD + c0]), v1 = bf2f(qkv[row + 2 * DD + c0 + 1]);

  float c = fcos[t * (HD / 2) + i];
  float s = fsin[t * (HD / 2) + i];

  float q0r = (q0 * c - q1 * s) * QSCALE;
  float q1r = (q0 * s + q1 * c) * QSCALE;
  float k0r = k0 * c - k1 * s;
  float k1r = k0 * s + k1 * c;

  size_t hb = ((size_t)(b * HH + h) * TT + t) * HD;
  qh[hb + 2 * i]     = f2bf(q0r);
  qh[hb + 2 * i + 1] = f2bf(q1r);
  kh[hb + 2 * i]     = f2bf(k0r);
  kh[hb + 2 * i + 1] = f2bf(k1r);

  size_t vb = (size_t)(b * HH + h) * HD;
  vT[(vb + 2 * i)     * TT + t] = f2bf(v0);
  vT[(vb + 2 * i + 1) * TT + t] = f2bf(v1);
}

// ---------------------------------------------------------------------------
// Flash attention. Block = 8 waves on one (b,h), 128 q rows (16/wave).
// K/V tiles (32 keys) are staged cooperatively into double-buffered LDS with
// global_load_async_to_lds_b128 (ASYNCcnt) so the copy of tile kt+1 overlaps
// the WMMA + softmax work on tile kt. All matrix math via bf16 WMMA.
// ---------------------------------------------------------------------------
__global__ __launch_bounds__(256) void flash_attn_kernel(
    const __bf16* __restrict__ qh, const __bf16* __restrict__ kh,
    const __bf16* __restrict__ vT, __bf16* __restrict__ attn) {
  __shared__ __align__(16) __bf16 Kbuf[2][32][HD];  // 2 x 4 KB
  __shared__ __align__(16) __bf16 Vbuf[2][HD][32];  // 2 x 4 KB
  __shared__ __align__(16) float  Slds[8][16][33];  // padded for row scans
  __shared__ __align__(16) __bf16 Plds[8][16][32];  // ld=32 keeps 16B align
  __shared__ float Alds[8][16];
  __shared__ float Llds[8][16];

  int tid  = threadIdx.x;
  int lane = tid & 31;
  int wid  = tid >> 5;
  int half = lane >> 4, col0 = lane & 15;

  int qtiles = TT / 128;                 // 16
  int bh = blockIdx.x / qtiles;
  int q0 = (blockIdx.x % qtiles) * 128 + wid * 16;
  int b = bh >> 4, h = bh & (HH - 1);

  const __bf16* qbase = qh + ((size_t)bh * TT) * HD;
  const __bf16* kbase = kh + ((size_t)bh * TT) * HD;
  const __bf16* vbase = vT + ((size_t)bh * HD) * TT;

  // per-thread async-copy coordinates (16 B per thread per tile)
  int kr = tid >> 3, kc = (tid & 7) << 4;      // K tile: 32 rows x 128 B
  int vr = tid >> 2, vc = (tid & 3) << 4;      // V tile: 64 rows x 64 B
  const char* kg = (const char*)(kbase + (size_t)kr * HD) + kc;
  const char* vg = (const char*)vbase + (size_t)vr * (TT * 2) + vc;
  unsigned kl0 = lds_off(&Kbuf[0][0][0]) + kr * (HD * 2) + kc;
  unsigned kl1 = lds_off(&Kbuf[1][0][0]) + kr * (HD * 2) + kc;
  unsigned vl0 = lds_off(&Vbuf[0][0][0]) + vr * 64 + vc;
  unsigned vl1 = lds_off(&Vbuf[1][0][0]) + vr * 64 + vc;

  auto issue_tile = [&](int kt, int buf) {
    size_t gk = (size_t)kt * 32 * (HD * 2);    // bytes into K rows
    size_t gv = (size_t)kt * 32 * 2;           // bytes into V cols
    async_ld_b128(buf ? kl1 : kl0, kg + gk);
    async_ld_b128(buf ? vl1 : vl0, vg + gv);
  };

  // Q fragments for rows q0..q0+15, K-dim = HD = 64 -> 2 chunks of 32
  v16bf aq[2];
#pragma unroll
  for (int c = 0; c < 2; ++c)
    aq[c] = load_frag(qbase + (size_t)q0 * HD + 32 * c, HD, lane);

  const v8f vzero = {0.f, 0.f, 0.f, 0.f, 0.f, 0.f, 0.f, 0.f};
  v8f o[4];
#pragma unroll
  for (int nb = 0; nb < 4; ++nb) o[nb] = vzero;

  float m_run = -1e30f, l_run = 0.f;     // meaningful in lanes 0..15

  issue_tile(0, 0);                      // prologue: tile 0 -> buffer 0

  const int NT = TT / 32;                // 64 key tiles
  for (int kt = 0; kt < NT; ++kt) {
    int cur = kt & 1;
    wait_async0();
    __syncthreads();                     // tile kt resident in Kbuf/Vbuf[cur]
    if (kt + 1 < NT) issue_tile(kt + 1, cur ^ 1);   // overlap next copy

    // ---- S = (q * scale * log2e) . K^T   (16 x 32 tile), K from LDS ----
    v8f s[2];
#pragma unroll
    for (int j = 0; j < 2; ++j) {
      v16bf bk0 = load_frag(&Kbuf[cur][j * 16][0], HD, lane);
      v16bf bk1 = load_frag(&Kbuf[cur][j * 16][0] + 32, HD, lane);
      v8f c = wmma_bf16(aq[0], bk0, vzero);
      s[j]  = wmma_bf16(aq[1], bk1, c);
    }

    // spill score tile to LDS in row-major form
#pragma unroll
    for (int j = 0; j < 2; ++j)
#pragma unroll
      for (int v = 0; v < 8; ++v)
        Slds[wid][v + 8 * half][j * 16 + col0] = s[j][v];
    __syncthreads();

    // ---- online softmax: lane r handles row r ----
    if (lane < 16) {
      float* srow = &Slds[wid][lane][0];
      float mt = srow[0];
#pragma unroll
      for (int j2 = 1; j2 < 32; ++j2) mt = fmaxf(mt, srow[j2]);
      float mnew  = fmaxf(m_run, mt);
      float alpha = exp2f(m_run - mnew);
      float sum = 0.f;
#pragma unroll
      for (int j2 = 0; j2 < 32; ++j2) {
        float p = exp2f(srow[j2] - mnew);
        sum += p;
        Plds[wid][lane][j2] = f2bf(p);
      }
      l_run = l_run * alpha + sum;
      m_run = mnew;
      Alds[wid][lane] = alpha;
    }
    __syncthreads();

    // ---- rescale accumulators by per-row alpha ----
#pragma unroll
    for (int v = 0; v < 8; ++v) {
      float al = Alds[wid][v + 8 * half];
#pragma unroll
      for (int nb = 0; nb < 4; ++nb) o[nb][v] *= al;
    }

    // ---- O += P . V  (K-dim = 32 keys, N = HD = 64), V from LDS ----
    v16bf ap = load_frag(&Plds[wid][0][0], 32, lane);
#pragma unroll
    for (int nb = 0; nb < 4; ++nb) {
      v16bf bv = load_frag(&Vbuf[cur][nb * 16][0], 32, lane);
      o[nb] = wmma_bf16(ap, bv, o[nb]);
    }
  }

  if (lane < 16) Llds[wid][lane] = l_run;
  __syncthreads();

  // ---- normalize and store attn[B,T,H*HD] as bf16 ----
#pragma unroll
  for (int v = 0; v < 8; ++v) {
    int m = v + 8 * half;
    float inv = 1.0f / Llds[wid][m];
    int t = q0 + m;
    size_t rowbase = ((size_t)(b * TT + t)) * DD + h * HD;
#pragma unroll
    for (int nb = 0; nb < 4; ++nb)
      attn[rowbase + nb * 16 + col0] = f2bf(o[nb][v] * inv);
  }
}

// ---------------------------------------------------------------------------
// Launch: convert -> GEMM1 (qkv) -> RoPE/split -> flash attention -> GEMM2
// ---------------------------------------------------------------------------
extern "C" void kernel_launch(void* const* d_in, const int* in_sizes, int n_in,
                              void* d_out, int out_size, void* d_ws,
                              size_t ws_size, hipStream_t stream) {
  const float* q      = (const float*)d_in[0];
  const float* fcos   = (const float*)d_in[1];
  const float* fsin   = (const float*)d_in[2];
  const float* qkv_w  = (const float*)d_in[3];
  const float* qkv_b  = (const float*)d_in[4];
  const float* o_w    = (const float*)d_in[5];
  const float* o_b    = (const float*)d_in[6];
  float* out = (float*)d_out;

  char* ws = (char*)d_ws;
  size_t off = 0;
  auto salloc = [&](size_t bytes) -> void* {
    void* p = ws + off;
    off = (off + bytes + 255) & ~(size_t)255;
    return p;
  };

  __bf16* q_bf    = (__bf16*)salloc((size_t)MM * DD * 2);        // 16 MB
  __bf16* w1_bf   = (__bf16*)salloc((size_t)N_QKV * DD * 2);     //  6 MB
  __bf16* w2_bf   = (__bf16*)salloc((size_t)DD * DD * 2);        //  2 MB
  __bf16* qkv_bf  = (__bf16*)salloc((size_t)MM * N_QKV * 2);     // 48 MB
  __bf16* qh      = (__bf16*)salloc((size_t)MM * DD * 2);        // 16 MB
  __bf16* khd     = (__bf16*)salloc((size_t)MM * DD * 2);        // 16 MB
  __bf16* vTd     = (__bf16*)salloc((size_t)MM * DD * 2);        // 16 MB
  __bf16* attn_bf = (__bf16*)salloc((size_t)MM * DD * 2);        // 16 MB

  // 1) fp32 -> bf16 converts
  {
    int n;
    n = MM * DD;
    f32_to_bf16_kernel<<<(n + 255) / 256, 256, 0, stream>>>(q, q_bf, n);
    n = N_QKV * DD;
    f32_to_bf16_kernel<<<(n + 255) / 256, 256, 0, stream>>>(qkv_w, w1_bf, n);
    n = DD * DD;
    f32_to_bf16_kernel<<<(n + 255) / 256, 256, 0, stream>>>(o_w, w2_bf, n);
  }

  // 2) qkv = q @ qkv_w^T + qkv_b   (bf16 out)
  {
    int totalWaves = (MM / 64) * (N_QKV / 32);     // 12288
    gemm_bf16_kernel<true><<<totalWaves / 8, 256, 0, stream>>>(
        q_bf, w1_bf, qkv_b, (void*)qkv_bf, MM, N_QKV);
  }

  // 3) RoPE + split + scale q + transpose v
  {
    int n = BB * TT * HH * (HD / 2);               // 2^22
    rope_split_kernel<<<n / 256, 256, 0, stream>>>(qkv_bf, fcos, fsin, qh, khd,
                                                   vTd);
  }

  // 4) flash attention (async LDS staged K/V)
  {
    int nblocks = BB * HH * (TT / 128);            // 1024
    flash_attn_kernel<<<nblocks, 256, 0, stream>>>(qh, khd, vTd, attn_bf);
  }

  // 5) out = attn @ o_w^T + o_b   (fp32 out)
  {
    int totalWaves = (MM / 64) * (DD / 32);        // 4096
    gemm_bf16_kernel<false><<<totalWaves / 8, 256, 0, stream>>>(
        attn_bf, w2_bf, o_b, (void*)out, MM, DD);
  }
}
